// RNN_69114613728698
// MI455X (gfx1250) — matrix operand order repaired and checked by reading, not compiled, not run on hardware
//
#include <hip/hip_runtime.h>
#include <hip/hip_bf16.h>
#include <stdint.h>

// Problem dims
#define B_  64
#define T_  256
#define D_  256
#define H_  512
#define C_  1000
#define KSPLIT 8

typedef __attribute__((ext_vector_type(16))) __bf16 v16bf;
typedef __attribute__((ext_vector_type(8)))  float  v8f;

union ABfrag { v16bf v; uint4 q[2]; };
union BVfrag { v16bf v; __bf16 e[16]; };

static __device__ __forceinline__ float fast_tanh(float x) {
  // tanh(x) = 1 - 2/(exp(2x)+1);  exp(2x) = exp2(x * 2*log2(e))
  float e = __builtin_amdgcn_exp2f(x * 2.8853900817779268f);
  return 1.0f - 2.0f * __builtin_amdgcn_rcpf(e + 1.0f);
}

// ---------------------------------------------------------------- cvt f32->bf16
__global__ void cvt_bf16_kernel(const float* __restrict__ src,
                                __bf16* __restrict__ dst, int n) {
  int i = blockIdx.x * blockDim.x + threadIdx.x;
  int stride = gridDim.x * blockDim.x;
  for (; i < n; i += stride) dst[i] = (__bf16)src[i];
}

// ---------------------------------------------------------------- xw GEMM
// out[M, H_] = A[M, K]_bf16 @ W[H_, K]_bf16^T + (bih + bhh)
// grid: (M/16, H_/64), block: 128 (4 waves, one 16-wide n-tile each)
template<int K>
__global__ __launch_bounds__(128)
void gemm_xw_kernel(const __bf16* __restrict__ A,
                    const __bf16* __restrict__ W,
                    const float* __restrict__ bih,
                    const float* __restrict__ bhh,
                    float* __restrict__ out) {
  const int lane = threadIdx.x & 31;
  const int wave = threadIdx.x >> 5;
  const int asel = lane >> 4;          // which half-wave (K sub-block select)
  const int mloc = lane & 15;
  const int ntile = blockIdx.y * 4 + wave;
  const int ncol  = ntile * 16 + mloc;
  const __bf16* arow = A + ((size_t)blockIdx.x * 16 + mloc) * K;
  const __bf16* wrow = W + (size_t)ncol * K;
  v8f c = {};
#pragma unroll
  for (int kt = 0; kt < K / 32; ++kt) {
    ABfrag a, b;
    const int ak = kt * 32 + asel * 16;    // A: lanes 0-15 K=0..15; 16-31 K=16..31
    a.q[0] = *(const uint4*)(arow + ak);
    a.q[1] = *(const uint4*)(arow + ak + 8);
    const int bk = kt * 32 + asel * 16;    // B: lanes 0-15 K=0..15; 16-31 K=16..31
    b.q[0] = *(const uint4*)(wrow + bk);
    b.q[1] = *(const uint4*)(wrow + bk + 8);
    c = __builtin_amdgcn_wmma_f32_16x16x32_bf16(false, a.v, false, b.v,
                                                (short)0, c, false, false);
  }
  const float bias = bih[ncol] + bhh[ncol];
  const size_t mbase = (size_t)blockIdx.x * 16 + asel * 8;
#pragma unroll
  for (int r = 0; r < 8; ++r)
    out[(mbase + r) * (size_t)H_ + ncol] = c[r] + bias;
}

// ---------------------------------------------------------------- recurrence
// h_t = tanh(xw[:,t,:] + h_{t-1} @ W_hh^T); per-block: 16 batches, all 512 cols.
// 32 waves; wave = n-tile. h_prev double-buffered in LDS (bf16).
// Register/stream split: 8 W_hh k-tiles resident in VGPRs (64 regs),
// 8 streamed from L2 each step. An opaque per-iteration integer offset
// (not a laundered pointer!) prevents hoist+spill while keeping the
// global address space so streams lower to global_load_b128, not flat.
#define KREG 8
__global__ __launch_bounds__(1024)
void rnn_rec_kernel(const float* __restrict__ xw,    // [B_,T_,H_], biases included
                    const __bf16* __restrict__ Whh,  // [H_,H_] bf16
                    __bf16* __restrict__ hout) {     // [B_,T_,H_] bf16
  __shared__ __bf16 hbuf[2][16][H_];                 // 2 x 16KB
  const int lane = threadIdx.x & 31;
  const int wave = threadIdx.x >> 5;                 // n-tile 0..31
  const int asel = lane >> 4;
  const int mloc = lane & 15;
  const int ncol = wave * 16 + mloc;
  const int mb   = blockIdx.x;                       // batch tile (16 rows)

  for (int i = threadIdx.x; i < 16 * H_; i += 1024)
    (&hbuf[0][0][0])[i] = (__bf16)0.0f;

  const __bf16* wrow = Whh + (size_t)ncol * H_;

  // Persistent B fragments: k-tiles 0..KREG-1 live in registers for all 256 steps.
  v16bf breg[KREG];
#pragma unroll
  for (int kt = 0; kt < KREG; ++kt) {
    ABfrag b;
    const int bk = kt * 32 + asel * 16;
    b.q[0] = *(const uint4*)(wrow + bk);
    b.q[1] = *(const uint4*)(wrow + bk + 8);
    breg[kt] = b.v;
  }
  __syncthreads();

  // xw software pipeline: preload t=0
  const size_t xstride = (size_t)T_ * H_;
  const float* xcol = xw + (((size_t)mb * 16 + asel * 8) * T_) * H_ + ncol;
  v8f xv;
#pragma unroll
  for (int r = 0; r < 8; ++r) xv[r] = xcol[(size_t)r * xstride];

  int p = 0;
  for (int t = 0; t < T_; ++t) {
    v8f c = xv;
    const __bf16* arow = &hbuf[p][mloc][0];

    // Phase 1: register-resident W_hh k-tiles
#pragma unroll
    for (int kt = 0; kt < KREG; ++kt) {
      ABfrag a;
      const int ak = kt * 32 + asel * 16;
      a.q[0] = *(const uint4*)(arow + ak);           // ds_load_b128
      a.q[1] = *(const uint4*)(arow + ak + 8);
      c = __builtin_amdgcn_wmma_f32_16x16x32_bf16(false, a.v, false, breg[kt],
                                                  (short)0, c, false, false);
    }

    // Phase 2: stream remaining k-tiles from L2. Opaque zero offset makes
    // the addresses loop-variant (no hoist/CSE) but keeps global addrspace.
    int stream_off = 0;
    asm volatile("" : "+v"(stream_off));
    const __bf16* wrow_t = wrow + stream_off;
#pragma unroll
    for (int kt = KREG; kt < H_ / 32; ++kt) {
      ABfrag a, b;
      const int ak = kt * 32 + asel * 16;
      a.q[0] = *(const uint4*)(arow + ak);
      a.q[1] = *(const uint4*)(arow + ak + 8);
      const int bk = kt * 32 + asel * 16;
      b.q[0] = *(const uint4*)(wrow_t + bk);         // global_load_b128
      b.q[1] = *(const uint4*)(wrow_t + bk + 8);
      c = __builtin_amdgcn_wmma_f32_16x16x32_bf16(false, a.v, false, b.v,
                                                  (short)0, c, false, false);
    }

    // Prefetch xw for t+1 (overlaps with tanh + stores + barrier)
    const int tn = (t + 1 < T_) ? (t + 1) : t;
#pragma unroll
    for (int r = 0; r < 8; ++r) xv[r] = xcol[(size_t)r * xstride + (size_t)tn * H_];

    // tanh + store (C layout: n = lane&15, m = r + 8*asel)
#pragma unroll
    for (int r = 0; r < 8; ++r) {
      const int m = asel * 8 + r;
      const __bf16 hv = (__bf16)fast_tanh(c[r]);
      hbuf[p ^ 1][m][ncol] = hv;
      hout[(((size_t)mb * 16 + m) * T_ + t) * H_ + ncol] = hv;
    }
    __syncthreads();   // single barrier per step thanks to ping-pong buffers
    p ^= 1;
  }
}

// ---------------------------------------------------------------- FC
static __device__ __forceinline__ v16bf load_bf16_from_f32(const float* p) {
  BVfrag b;
  const float4* q = (const float4*)p;
  float4 f0 = q[0], f1 = q[1], f2 = q[2], f3 = q[3];
  b.e[0]  = (__bf16)f0.x; b.e[1]  = (__bf16)f0.y; b.e[2]  = (__bf16)f0.z; b.e[3]  = (__bf16)f0.w;
  b.e[4]  = (__bf16)f1.x; b.e[5]  = (__bf16)f1.y; b.e[6]  = (__bf16)f1.z; b.e[7]  = (__bf16)f1.w;
  b.e[8]  = (__bf16)f2.x; b.e[9]  = (__bf16)f2.y; b.e[10] = (__bf16)f2.z; b.e[11] = (__bf16)f2.w;
  b.e[12] = (__bf16)f3.x; b.e[13] = (__bf16)f3.y; b.e[14] = (__bf16)f3.z; b.e[15] = (__bf16)f3.w;
  return b.v;
}

// out_part[ks][B_][C_] partial GEMM: A = h1 [B_, K], W = fc_w [C_, K] f32 (cvt inline)
// grid: (ceil(C_/16)=63, KSPLIT), block 128: wave = m-tile (4 x 16 = all 64 rows)
__global__ __launch_bounds__(128)
void fc_partial_kernel(const __bf16* __restrict__ A,
                       const float* __restrict__ Wfc,
                       float* __restrict__ part) {
  const int K = T_ * H_;                 // 131072
  const int lane = threadIdx.x & 31;
  const int wave = threadIdx.x >> 5;     // m-tile
  const int asel = lane >> 4;
  const int mloc = lane & 15;
  const int ncol = blockIdx.x * 16 + mloc;            // 0..1007
  const int ncl  = ncol < C_ ? ncol : (C_ - 1);       // clamp row for loads
  const int k0   = blockIdx.y * (K / KSPLIT);
  const __bf16* arow = A + (size_t)(wave * 16 + mloc) * K + k0;
  const float*  wrow = Wfc + (size_t)ncl * K + k0;

  v8f c0 = {}, c1 = {};
  const int KT = (K / KSPLIT) / 32;      // 512
  for (int kt = 0; kt < KT; kt += 2) {
    {
      ABfrag a;
      const int ak = kt * 32 + asel * 16;
      a.q[0] = *(const uint4*)(arow + ak);
      a.q[1] = *(const uint4*)(arow + ak + 8);
      v16bf bv = load_bf16_from_f32(wrow + kt * 32 + asel * 16);
      c0 = __builtin_amdgcn_wmma_f32_16x16x32_bf16(false, a.v, false, bv,
                                                   (short)0, c0, false, false);
    }
    {
      ABfrag a;
      const int ak = (kt + 1) * 32 + asel * 16;
      a.q[0] = *(const uint4*)(arow + ak);
      a.q[1] = *(const uint4*)(arow + ak + 8);
      v16bf bv = load_bf16_from_f32(wrow + (kt + 1) * 32 + asel * 16);
      c1 = __builtin_amdgcn_wmma_f32_16x16x32_bf16(false, a.v, false, bv,
                                                   (short)0, c1, false, false);
    }
  }
  v8f cs = c0 + c1;
  if (ncol < C_) {
#pragma unroll
    for (int r = 0; r < 8; ++r) {
      const int brow = wave * 16 + asel * 8 + r;
      part[((size_t)blockIdx.y * B_ + brow) * C_ + ncol] = cs[r];
    }
  }
}

__global__ void fc_reduce_kernel(const float* __restrict__ part,
                                 const float* __restrict__ fcb,
                                 float* __restrict__ out) {
  int i = blockIdx.x * blockDim.x + threadIdx.x;
  if (i >= B_ * C_) return;
  const int b = i / C_, c = i % C_;
  float s = fcb[c];
#pragma unroll
  for (int ks = 0; ks < KSPLIT; ++ks)
    s += part[((size_t)ks * B_ + b) * C_ + c];
  out[i] = s;
}

// ---------------------------------------------------------------- launch
extern "C" void kernel_launch(void* const* d_in, const int* in_sizes, int n_in,
                              void* d_out, int out_size, void* d_ws, size_t ws_size,
                              hipStream_t stream) {
  const float* inp   = (const float*)d_in[0];
  const float* W_ih0 = (const float*)d_in[1];
  const float* W_hh0 = (const float*)d_in[2];
  const float* b_ih0 = (const float*)d_in[3];
  const float* b_hh0 = (const float*)d_in[4];
  const float* W_ih1 = (const float*)d_in[5];
  const float* W_hh1 = (const float*)d_in[6];
  const float* b_ih1 = (const float*)d_in[7];
  const float* b_hh1 = (const float*)d_in[8];
  const float* fc_w  = (const float*)d_in[9];
  const float* fc_b  = (const float*)d_in[10];

  char* ws = (char*)d_ws;
  size_t off = 0;
  auto alloc = [&](size_t bytes) -> void* {
    void* p = ws + off;
    off = (off + bytes + 255) & ~(size_t)255;
    return p;
  };
  __bf16* inp_bf = (__bf16*)alloc((size_t)B_ * T_ * D_ * 2);
  __bf16* wih0   = (__bf16*)alloc((size_t)H_ * D_ * 2);
  __bf16* whh0   = (__bf16*)alloc((size_t)H_ * H_ * 2);
  __bf16* wih1   = (__bf16*)alloc((size_t)H_ * H_ * 2);
  __bf16* whh1   = (__bf16*)alloc((size_t)H_ * H_ * 2);
  float*  xw     = (float*)alloc((size_t)B_ * T_ * H_ * 4);
  __bf16* h0     = (__bf16*)alloc((size_t)B_ * T_ * H_ * 2);
  __bf16* h1     = (__bf16*)alloc((size_t)B_ * T_ * H_ * 2);
  float*  part   = (float*)alloc((size_t)KSPLIT * B_ * C_ * 4);

  // f32 -> bf16 conversions
  cvt_bf16_kernel<<<1024, 256, 0, stream>>>(inp,   inp_bf, B_ * T_ * D_);
  cvt_bf16_kernel<<<256,  256, 0, stream>>>(W_ih0, wih0,   H_ * D_);
  cvt_bf16_kernel<<<256,  256, 0, stream>>>(W_hh0, whh0,   H_ * H_);
  cvt_bf16_kernel<<<256,  256, 0, stream>>>(W_ih1, wih1,   H_ * H_);
  cvt_bf16_kernel<<<256,  256, 0, stream>>>(W_hh1, whh1,   H_ * H_);

  // layer 0
  gemm_xw_kernel<D_><<<dim3(B_ * T_ / 16, H_ / 64), 128, 0, stream>>>(
      inp_bf, wih0, b_ih0, b_hh0, xw);
  rnn_rec_kernel<<<B_ / 16, 1024, 0, stream>>>(xw, whh0, h0);

  // layer 1
  gemm_xw_kernel<H_><<<dim3(B_ * T_ / 16, H_ / 64), 128, 0, stream>>>(
      h0, wih1, b_ih1, b_hh1, xw);
  rnn_rec_kernel<<<B_ / 16, 1024, 0, stream>>>(xw, whh1, h1);

  // FC: stream fc_w (524MB) once from HBM, WMMA with inline f32->bf16
  fc_partial_kernel<<<dim3((C_ + 15) / 16, KSPLIT), 128, 0, stream>>>(h1, fc_w, part);
  fc_reduce_kernel<<<(B_ * C_ + 255) / 256, 256, 0, stream>>>(part, fc_b, (float*)d_out);
}
